// ICR_19868518711721
// MI455X (gfx1250) — matrix operand order repaired and checked by reading, not compiled, not run on hardware
//
#include <hip/hip_runtime.h>
#include <hip/hip_bf16.h>
#include <math.h>

typedef __attribute__((ext_vector_type(16))) _Float16 v16h;
typedef __attribute__((ext_vector_type(8)))  float    v8f;

#define PP 2048   // proposals per batch (fixed by setup)
#define EPS_F 1e-7f

// ---------------------------------------------------------------------------
// Kernel 0: zero the atomic accumulators (imbalance[4] in ws, loss scalar in d_out)
// ---------------------------------------------------------------------------
__global__ void icr_zero_kernel(float* __restrict__ imb, float* __restrict__ loss_out) {
    if (threadIdx.x < 4) imb[threadIdx.x] = 0.0f;
    if (threadIdx.x == 0) *loss_out = 0.0f;
}

// ---------------------------------------------------------------------------
// Kernel 1: per-(b,c) seed statistics: cnt = #(score>0.5), argmax (first index)
// grid = (B, 3), block = 256
// ---------------------------------------------------------------------------
__global__ void icr_seed_stats_kernel(const float* __restrict__ pre_score,
                                      int* __restrict__ cnt_arg /* [B*3*2] */) {
    const int b = blockIdx.x, c = blockIdx.y;
    __shared__ float sv[256];
    __shared__ int   si[256];
    __shared__ int   sc[256];
    float best = -1e30f; int bi = 0; int cnt = 0;
    for (int j = threadIdx.x; j < PP; j += 256) {
        float v = pre_score[((size_t)(b * PP + j)) * 4 + c];
        if (v > best || (v == best && j < bi)) { best = v; bi = j; }
        cnt += (v > 0.5f) ? 1 : 0;
    }
    sv[threadIdx.x] = best; si[threadIdx.x] = bi; sc[threadIdx.x] = cnt;
    __syncthreads();
    for (int s = 128; s > 0; s >>= 1) {
        if (threadIdx.x < s) {
            float ov = sv[threadIdx.x + s]; int oi = si[threadIdx.x + s];
            if (ov > sv[threadIdx.x] || (ov == sv[threadIdx.x] && oi < si[threadIdx.x])) {
                sv[threadIdx.x] = ov; si[threadIdx.x] = oi;
            }
            sc[threadIdx.x] += sc[threadIdx.x + s];
        }
        __syncthreads();
    }
    if (threadIdx.x == 0) {
        cnt_arg[(b * 3 + c) * 2 + 0] = sc[0];
        cnt_arg[(b * 3 + c) * 2 + 1] = si[0];
    }
}

// ---------------------------------------------------------------------------
// Kernel 2: xr = X @ W^T + bias via v_wmma_f32_16x16x32_f16, split-fp16 (hi+lo)
// for ~fp32 accuracy: X*W ~= Xh*Wh + Xh*Wl + Xl*Wh (all f32-accumulated).
// Each wave produces a 16-row x 16-col tile (cols 0..3 valid). 4 waves/block.
// ---------------------------------------------------------------------------
__global__ void icr_gemm_wmma_kernel(const float* __restrict__ X,
                                     const float* __restrict__ W,   // (4, C)
                                     const float* __restrict__ bias,
                                     float* __restrict__ xr,        // (rows, 4)
                                     int C) {
    const int wave    = threadIdx.x >> 5;
    const int lane    = threadIdx.x & 31;
    const int laneMod = lane & 15;
    const int halfSel = lane >> 4;                 // 0: lanes 0-15, 1: lanes 16-31
    const int rowBase = (blockIdx.x * 4 + wave) * 16;

    v8f acc = {};
    const int kb = halfSel * 8;                    // A-layout K sub-offset per half-wave
    for (int k0 = 0; k0 < C; k0 += 32) {
        // ---- A fragment (16x32 f16): row = rowBase + laneMod
        const float* arow = X + (size_t)(rowBase + laneMod) * C + k0;
        v16h ah, al;
#pragma unroll
        for (int e = 0; e < 8; ++e) {
            float x = arow[kb + e];
            _Float16 h = (_Float16)x;
            ah[e] = h; al[e] = (_Float16)(x - (float)h);
        }
#pragma unroll
        for (int e = 0; e < 8; ++e) {
            float x = arow[16 + kb + e];
            _Float16 h = (_Float16)x;
            ah[8 + e] = h; al[8 + e] = (_Float16)(x - (float)h);
        }
        // ---- B fragment (32x16 f16): N = laneMod, K = halfSel*8 + e (e<8), 16 + halfSel*8 + e-8 (e>=8)
        v16h bh, bl;
        if (laneMod < 4) {
            const float* wrow = W + (size_t)laneMod * C + k0 + halfSel * 8;
#pragma unroll
            for (int e = 0; e < 16; ++e) {
                float x = wrow[e < 8 ? e : e + 8];
                _Float16 h = (_Float16)x;
                bh[e] = h; bl[e] = (_Float16)(x - (float)h);
            }
        } else {
#pragma unroll
            for (int e = 0; e < 16; ++e) { bh[e] = (_Float16)0.0f; bl[e] = (_Float16)0.0f; }
        }
        // D += Ah*Bh + Ah*Bl + Al*Bh   (f32 accumulate)
        acc = __builtin_amdgcn_wmma_f32_16x16x32_f16(false, ah, false, bh, (short)0, acc, false, false);
        acc = __builtin_amdgcn_wmma_f32_16x16x32_f16(false, ah, false, bl, (short)0, acc, false, false);
        acc = __builtin_amdgcn_wmma_f32_16x16x32_f16(false, al, false, bh, (short)0, acc, false, false);
    }
    // D layout: element r -> row rowBase + r + halfSel*8, col laneMod
    if (laneMod < 4) {
        float bsv = bias[laneMod];
#pragma unroll
        for (int r = 0; r < 8; ++r) {
            int row = rowBase + r + halfSel * 8;
            xr[(size_t)row * 4 + laneMod] = acc[r] + bsv;
        }
    }
}

// ---------------------------------------------------------------------------
// Kernel 3: assignment. grid = (P/256, B), block = 256. All rois[b] + scores[b]
// staged in LDS (32KB + 24KB + 2KB seed mask << 320KB/WGP). Each thread owns
// one proposal i and streams j = 0..P-1 from LDS (broadcast reads).
// ---------------------------------------------------------------------------
__global__ void icr_assign_kernel(const float* __restrict__ rois,      // (B,P,4)
                                  const float* __restrict__ pre_score, // (B,P,4)
                                  const float* __restrict__ labels,    // (B,4)
                                  const int*   __restrict__ cnt_arg,   // (B,3,2)
                                  float* __restrict__ y_out,           // (B*P,4)
                                  float* __restrict__ w_out,           // (B*P)
                                  float* __restrict__ imb)             // (4)
{
    const int b  = blockIdx.y;
    const int i  = blockIdx.x * 256 + threadIdx.x;

    __shared__ float          s_roi[PP * 4];
    __shared__ float          s_sc [PP * 3];
    __shared__ unsigned char  s_seed[PP];

    int cnt_c[3], arg_c[3];
#pragma unroll
    for (int c = 0; c < 3; ++c) {
        cnt_c[c] = cnt_arg[(b * 3 + c) * 2 + 0];
        arg_c[c] = cnt_arg[(b * 3 + c) * 2 + 1];
    }

    for (int j = threadIdx.x; j < PP; j += 256) {
        const float4 r4 = ((const float4*)rois)[(size_t)b * PP + j];
        s_roi[j * 4 + 0] = r4.x; s_roi[j * 4 + 1] = r4.y;
        s_roi[j * 4 + 2] = r4.z; s_roi[j * 4 + 3] = r4.w;
        const float4 p4 = ((const float4*)pre_score)[(size_t)b * PP + j];
        float scv[3] = {p4.x, p4.y, p4.z};
        s_sc[j * 3 + 0] = p4.x; s_sc[j * 3 + 1] = p4.y; s_sc[j * 3 + 2] = p4.z;
        unsigned char m = 0;
#pragma unroll
        for (int c = 0; c < 3; ++c) {
            bool seed = (cnt_c[c] <= 1) ? (j == arg_c[c]) : (scv[c] > 0.5f);
            m |= (unsigned char)(seed ? (1 << c) : 0);
        }
        s_seed[j] = m;
    }
    __syncthreads();

    const float4 ri = ((const float4*)rois)[(size_t)b * PP + i];
    const float areai = (ri.z - ri.x) * (ri.w - ri.y);

    float best[3] = {-1.0f, -1.0f, -1.0f};
    int   bj[3]   = {0, 0, 0};
    for (int j = 0; j < PP; ++j) {
        float x1 = s_roi[j * 4 + 0], y1 = s_roi[j * 4 + 1];
        float x2 = s_roi[j * 4 + 2], y2 = s_roi[j * 4 + 3];
        float lw = fmaxf(fminf(ri.z, x2) - fmaxf(ri.x, x1), 0.0f);
        float lh = fmaxf(fminf(ri.w, y2) - fmaxf(ri.y, y1), 0.0f);
        float inter = lw * lh;
        float areaj = (x2 - x1) * (y2 - y1);
        float iou   = inter / (areai + areaj - inter);
        unsigned char m = s_seed[j];
        // strict '>' keeps the first argmax occurrence, matching jnp.argmax
        if ((m & 1) && iou > best[0]) { best[0] = iou; bj[0] = j; }
        if ((m & 2) && iou > best[1]) { best[1] = iou; bj[1] = j; }
        if ((m & 4) && iou > best[2]) { best[2] = iou; bj[2] = j; }
    }

    float Iv = -1.0f, wv = 1.0f; int ycls = 3;
#pragma unroll
    for (int c = 0; c < 3; ++c) {
        bool upd = (labels[b * 4 + c] > 0.0f) && (best[c] >= 0.5f) && (best[c] > Iv);
        if (upd) { Iv = best[c]; wv = s_sc[bj[c] * 3 + c]; ycls = c; }
    }

    const size_t r = (size_t)b * PP + i;
    float4 yv = {0.0f, 0.0f, 0.0f, 0.0f};
    ((float*)&yv)[ycls] = 1.0f;
    ((float4*)y_out)[r] = yv;
    w_out[r] = wv;
    atomicAdd(&imb[ycls], 1.0f);
}

// ---------------------------------------------------------------------------
// Kernel 4: softmax -> logit (to d_out), focal loss, weighting, scalar reduce.
// grid = rows/256, block = 256.
// ---------------------------------------------------------------------------
__global__ void icr_loss_kernel(const float* __restrict__ xr,
                                const float* __restrict__ y,
                                const float* __restrict__ w,
                                const float* __restrict__ labels,
                                const float* __restrict__ imb,
                                float* __restrict__ logit_out,
                                float* __restrict__ loss_out,
                                float invB) {
    const int r = blockIdx.x * 256 + threadIdx.x;
    const int b = r / PP;

    const float4 v  = ((const float4*)xr)[r];
    float vm = fmaxf(fmaxf(v.x, v.y), fmaxf(v.z, v.w));
    float e0 = expf(v.x - vm), e1 = expf(v.y - vm), e2 = expf(v.z - vm), e3 = expf(v.w - vm);
    float inv = 1.0f / (e0 + e1 + e2 + e3);
    float lg[4] = {e0 * inv, e1 * inv, e2 * inv, e3 * inv};
    float4 lo = {lg[0], lg[1], lg[2], lg[3]};
    ((float4*)logit_out)[r] = lo;

    const float4 yv4 = ((const float4*)y)[r];
    const float  yv[4] = {yv4.x, yv4.y, yv4.z, yv4.w};
    const float  wr = w[r];

    float acc = 0.0f;
#pragma unroll
    for (int c = 0; c < 4; ++c) {
        float p  = fminf(fmaxf(lg[c], EPS_F), 1.0f - EPS_F);
        float om = 1.0f - p;
        float fl = -yv[c] * logf(p) * om * om;          // GAMMA = 2
        float ls = wr * fl / (imb[c] + EPS_F);
        float lab = (c == 3) ? 1.0f : labels[b * 4 + c];
        float w_  = 10.0f * expf(lg[c]) * (1.0f - lab) + lab;
        acc += ls * w_;
    }

    __shared__ float red[256];
    red[threadIdx.x] = acc;
    __syncthreads();
    for (int s = 128; s > 0; s >>= 1) {
        if (threadIdx.x < s) red[threadIdx.x] += red[threadIdx.x + s];
        __syncthreads();
    }
    if (threadIdx.x == 0) atomicAdd(loss_out, red[0] * invB);
}

// ---------------------------------------------------------------------------
extern "C" void kernel_launch(void* const* d_in, const int* in_sizes, int n_in,
                              void* d_out, int out_size, void* d_ws, size_t ws_size,
                              hipStream_t stream) {
    const float* inputs    = (const float*)d_in[0];  // (B*P, C)
    const float* pre_score = (const float*)d_in[1];  // (B, P, 4)
    const float* labels    = (const float*)d_in[2];  // (B, 4)
    const float* rois      = (const float*)d_in[3];  // (B, P, 4)
    const float* fc_weight = (const float*)d_in[4];  // (4, C)
    const float* fc_bias   = (const float*)d_in[5];  // (4,)

    const int B    = in_sizes[2] / 4;            // 16
    const int rows = in_sizes[1] / 4;            // B*P = 32768
    const int P    = rows / B;                   // 2048
    const int C    = in_sizes[0] / rows;         // 512
    (void)P;                                     // kernels assume P == PP (2048)

    // workspace layout (floats)
    float* ws_f   = (float*)d_ws;
    float* xr     = ws_f;                        // rows*4
    float* y_buf  = ws_f + (size_t)rows * 4;     // rows*4
    float* w_buf  = y_buf + (size_t)rows * 4;    // rows
    float* imb    = w_buf + rows;                // 4
    int*   cntArg = (int*)(imb + 4);             // B*3*2 ints

    float* logit_out = (float*)d_out;            // rows*4
    float* loss_out  = logit_out + (size_t)rows * 4;  // 1 scalar

    icr_zero_kernel<<<1, 32, 0, stream>>>(imb, loss_out);

    dim3 gStats(B, 3);
    icr_seed_stats_kernel<<<gStats, 256, 0, stream>>>(pre_score, cntArg);

    icr_gemm_wmma_kernel<<<rows / 64, 128, 0, stream>>>(inputs, fc_weight, fc_bias, xr, C);

    dim3 gAssign(PP / 256, B);
    icr_assign_kernel<<<gAssign, 256, 0, stream>>>(rois, pre_score, labels, cntArg,
                                                   y_buf, w_buf, imb);

    icr_loss_kernel<<<rows / 256, 256, 0, stream>>>(xr, y_buf, w_buf, labels, imb,
                                                    logit_out, loss_out, 1.0f / (float)B);
}
